// QwenEnhancedDrugSynergyModel_21749714387668
// MI455X (gfx1250) — matrix operand order repaired and checked by reading, not compiled, not run on hardware
//
#include <hip/hip_runtime.h>
#include <hip/hip_bf16.h>
#include <float.h>

typedef __bf16 bf16_t;
typedef __attribute__((ext_vector_type(16))) __bf16 v16bf;
typedef __attribute__((ext_vector_type(8)))  __bf16 v8bf;
typedef __attribute__((ext_vector_type(8)))  float  v8f;

__device__ inline bf16_t f2bf(float f) {
  unsigned u = __float_as_uint(f);
  unsigned r = (u + 0x7FFFu + ((u >> 16) & 1u)) >> 16;
  unsigned short s = (unsigned short)r;
  return __builtin_bit_cast(bf16_t, s);
}

// pack two floats into two bf16 (RNE) in one dword
__device__ inline unsigned pack2bf(float a, float b) {
  unsigned ua = __float_as_uint(a);
  unsigned ra = (ua + 0x7FFFu + ((ua >> 16) & 1u)) >> 16;
  unsigned ub = __float_as_uint(b);
  unsigned rb = (ub + 0x7FFFu + ((ub >> 16) & 1u)) >> 16;
  return (ra & 0xFFFFu) | (rb << 16);
}

// ---------------------------------------------------------------------------
// Generic row-major GEMM + bias: C[M,N] = A[M,K] @ W[K,N] + bias[N]
// bf16 WMMA, f32 accumulation. Block: 256 threads = 8 waves.
// Block tile 128x64, K-step 32; wave computes 32x32 (4 v_wmma per K-step).
// Hot loop: 8x ds_load_b128 + 4x v_wmma_f32_16x16x32_bf16 per wave per K-step.
// Staging: uniform-branch fast path (float2 global loads, packed LDS stores)
// for interior tiles; clamped+select scalar path only at edges/K-tail.
// ---------------------------------------------------------------------------
#define BM 128
#define BN 64
#define BK 32

__global__ __launch_bounds__(256)
void gemm_bias_wmma(const float* __restrict__ A, const float* __restrict__ W,
                    const float* __restrict__ bias, float* __restrict__ C,
                    int M, int K, int N) {
  __shared__ bf16_t As[BM][BK];        // A rows, row stride 64B (16B aligned)
  __shared__ bf16_t Bs[BN][BK + 8];    // B transposed (n,k), row stride 80B (16B aligned)

  const int t    = threadIdx.x;
  const int lane = t & 31;
  const int wave = t >> 5;
  const int bm   = blockIdx.y * BM;
  const int bn   = blockIdx.x * BN;
  const int wm   = (wave & 3) * 32;
  const int wn   = (wave >> 2) * 32;

  const bool aFull = (bm + BM <= M);   // uniform per block
  const bool bFull = (bn + BN <= N);   // uniform per block

  v8f c00 = {}, c01 = {}, c10 = {}, c11 = {};

  const int mrow0 = wm + (lane & 15);
  const int mrow1 = mrow0 + 16;
  const int hiA   = (lane & 16) ? 8 : 0;    // A lane halves: K 0-7/16-23 vs 8-15/24-31
  const int hiB   = (lane & 16) ? 8 : 0;    // B lane halves: K 0-7/16-23 vs 8-15/24-31
  const int nc0   = wn + (lane & 15);
  const int nc1   = nc0 + 16;

  for (int k0 = 0; k0 < K; k0 += BK) {
    const bool kFull = (k0 + BK <= K);      // uniform per iteration

    // prefetch next K-tile (speculative, lowers to global_prefetch_b8)
    if (k0 + BK < K) {
      int pr = bm + (t >> 1); pr = pr < M ? pr : M - 1;
      int pk = k0 + BK + ((t & 1) << 4); pk = pk < K ? pk : K - 1;
      __builtin_prefetch(&A[(long)pr * K + pk], 0, 1);
      int wr = k0 + BK + (t >> 3); wr = wr < K ? wr : K - 1;
      int wc = bn + ((t & 7) << 3); wc = wc < N ? wc : N - 1;
      __builtin_prefetch(&W[(long)wr * N + wc], 0, 1);
    }

    // ---- stage A tile (128x32)
    if (aFull && kFull) {
      // K is even for all layers => float2 loads are 8B aligned
      #pragma unroll
      for (int j = 0; j < (BM * BK) / 512; ++j) {          // 8 float2 per thread
        int idx = t + j * 256;
        int r = idx >> 4, c2 = (idx & 15) << 1;
        const float2 v = *(const float2*)&A[(long)(bm + r) * K + (k0 + c2)];
        *(unsigned*)&As[r][c2] = pack2bf(v.x, v.y);
      }
    } else {
      #pragma unroll
      for (int i = 0; i < (BM * BK) / 256; ++i) {
        int idx = t + i * 256;
        int r = idx >> 5, cc = idx & 31;
        int gr = bm + r, gk = k0 + cc;
        int grc = gr < M ? gr : M - 1;
        int gkc = gk < K ? gk : K - 1;
        float v = A[(long)grc * K + gkc];
        As[r][cc] = f2bf((gr < M && gk < K) ? v : 0.0f);
      }
    }
    // ---- stage B tile (32x64) transposed into Bs[n][k]
    if (bFull && kFull) {
      #pragma unroll
      for (int j = 0; j < (BK * BN) / 512; ++j) {          // 4 float2 per thread
        int idx = t + j * 256;
        int r = idx >> 5, c2 = (idx & 31) << 1;            // r = k, c2 = n
        const float2 v = *(const float2*)&W[(long)(k0 + r) * N + (bn + c2)];
        Bs[c2][r]     = f2bf(v.x);
        Bs[c2 + 1][r] = f2bf(v.y);
      }
    } else {
      #pragma unroll
      for (int i = 0; i < (BK * BN) / 256; ++i) {
        int idx = t + i * 256;
        int r = idx >> 6, cc = idx & 63;                   // r = k, cc = n
        int gk = k0 + r, gc = bn + cc;
        int gkc = gk < K ? gk : K - 1;
        int gcc = gc < N ? gc : N - 1;
        float v = W[(long)gkc * N + gcc];
        Bs[cc][r] = f2bf((gk < K && gc < N) ? v : 0.0f);
      }
    }
    __syncthreads();

    // ---- vectorized fragment fetch (b128 DS loads)
    v8bf a0lo = *(const v8bf*)&As[mrow0][hiA];
    v8bf a0hi = *(const v8bf*)&As[mrow0][hiA + 16];
    v8bf a1lo = *(const v8bf*)&As[mrow1][hiA];
    v8bf a1hi = *(const v8bf*)&As[mrow1][hiA + 16];
    v8bf b0lo = *(const v8bf*)&Bs[nc0][hiB];
    v8bf b0hi = *(const v8bf*)&Bs[nc0][hiB + 16];
    v8bf b1lo = *(const v8bf*)&Bs[nc1][hiB];
    v8bf b1hi = *(const v8bf*)&Bs[nc1][hiB + 16];
    v16bf a0 = __builtin_shufflevector(a0lo, a0hi, 0,1,2,3,4,5,6,7,8,9,10,11,12,13,14,15);
    v16bf a1 = __builtin_shufflevector(a1lo, a1hi, 0,1,2,3,4,5,6,7,8,9,10,11,12,13,14,15);
    v16bf b0 = __builtin_shufflevector(b0lo, b0hi, 0,1,2,3,4,5,6,7,8,9,10,11,12,13,14,15);
    v16bf b1 = __builtin_shufflevector(b1lo, b1hi, 0,1,2,3,4,5,6,7,8,9,10,11,12,13,14,15);

    c00 = __builtin_amdgcn_wmma_f32_16x16x32_bf16(false, a0, false, b0, (short)0, c00, false, false);
    c01 = __builtin_amdgcn_wmma_f32_16x16x32_bf16(false, a0, false, b1, (short)0, c01, false, false);
    c10 = __builtin_amdgcn_wmma_f32_16x16x32_bf16(false, a1, false, b0, (short)0, c10, false, false);
    c11 = __builtin_amdgcn_wmma_f32_16x16x32_bf16(false, a1, false, b1, (short)0, c11, false, false);
    __syncthreads();
  }

  // C/D layout: VGPR r -> M = r (lanes 0-15) or r+8 (lanes 16-31); N = lane&15
  const int row0 = bm + wm + ((lane & 16) ? 8 : 0);
  const int row1 = row0 + 16;
  const int col0 = bn + wn + (lane & 15);
  const int col1 = col0 + 16;
  const float bia0 = (col0 < N) ? bias[col0] : 0.0f;
  const float bia1 = (col1 < N) ? bias[col1] : 0.0f;
  #pragma unroll
  for (int r = 0; r < 8; ++r) {
    if (row0 + r < M) {
      if (col0 < N) C[(long)(row0 + r) * N + col0] = c00[r] + bia0;
      if (col1 < N) C[(long)(row0 + r) * N + col1] = c01[r] + bia1;
    }
    if (row1 + r < M) {
      if (col0 < N) C[(long)(row1 + r) * N + col0] = c10[r] + bia0;
      if (col1 < N) C[(long)(row1 + r) * N + col1] = c11[r] + bia1;
    }
  }
}

// ---------------------------------------------------------------------------
// Elementwise / graph kernels
// ---------------------------------------------------------------------------
__global__ void fill_kernel(float* __restrict__ p, long n, float v) {
  long i = (long)blockIdx.x * blockDim.x + threadIdx.x;
  if (i < n) p[i] = v;
}

__device__ inline void atomicMaxF(float* addr, float v) {
  if (v >= 0.0f) atomicMax((int*)addr, __float_as_int(v));
  else           atomicMin((unsigned int*)addr, __float_as_uint(v));
}

// logit[e,h] = sum_c att[h,c] * leaky_relu(xl[src,h,c] + xr[dst,h,c], 0.2)
__global__ void edge_logit_kernel(const float* __restrict__ xl, const float* __restrict__ xr,
                                  const float* __restrict__ att,
                                  const int* __restrict__ src, const int* __restrict__ dst,
                                  int E, int Nn, int H, int Cc, float* __restrict__ logit) {
  long i = (long)blockIdx.x * blockDim.x + threadIdx.x;
  long total = (long)(E + Nn) * H;
  if (i >= total) return;
  int h = (int)(i % H); long e = i / H;
  int s = (e < E) ? src[e] : (int)(e - E);
  int d = (e < E) ? dst[e] : (int)(e - E);
  const float* pl = xl + (long)s * H * Cc + (long)h * Cc;
  const float* pr = xr + (long)d * H * Cc + (long)h * Cc;
  const float* pa = att + (long)h * Cc;
  float acc = 0.0f;
  for (int c = 0; c < Cc; ++c) {
    float v = pl[c] + pr[c];
    v = (v > 0.0f) ? v : 0.2f * v;
    acc += v * pa[c];
  }
  logit[e * H + h] = acc;
}

__global__ void edge_max_kernel(const float* __restrict__ logit,
                                const int* __restrict__ dst,
                                int E, int Nn, int H, float* __restrict__ segmax) {
  long i = (long)blockIdx.x * blockDim.x + threadIdx.x;
  long total = (long)(E + Nn) * H;
  if (i >= total) return;
  int h = (int)(i % H); long e = i / H;
  int d = (e < E) ? dst[e] : (int)(e - E);
  atomicMaxF(&segmax[(long)d * H + h], logit[i]);
}

__global__ void edge_exp_kernel(float* __restrict__ logit,
                                const int* __restrict__ dst,
                                int E, int Nn, int H,
                                const float* __restrict__ segmax, float* __restrict__ segsum) {
  long i = (long)blockIdx.x * blockDim.x + threadIdx.x;
  long total = (long)(E + Nn) * H;
  if (i >= total) return;
  int h = (int)(i % H); long e = i / H;
  int d = (e < E) ? dst[e] : (int)(e - E);
  float a = expf(logit[i] - segmax[(long)d * H + h]);
  logit[i] = a;
  atomicAdd(&segsum[(long)d * H + h], a);
}

// acc[dst,h,c] += (a / (segsum[dst,h] + 1e-16)) * xl[src,h,c]
__global__ void edge_aggregate_kernel(const float* __restrict__ alpha, const float* __restrict__ xl,
                                      const int* __restrict__ src, const int* __restrict__ dst,
                                      int E, int Nn, int H, int Cc,
                                      const float* __restrict__ segsum, float* __restrict__ acc) {
  long i = (long)blockIdx.x * blockDim.x + threadIdx.x;
  long total = (long)(E + Nn) * H;
  if (i >= total) return;
  int h = (int)(i % H); long e = i / H;
  int s = (e < E) ? src[e] : (int)(e - E);
  int d = (e < E) ? dst[e] : (int)(e - E);
  float w = alpha[i] / (segsum[(long)d * H + h] + 1e-16f);
  const float* pl = xl + (long)s * H * Cc + (long)h * Cc;
  float* po = acc + (long)d * H * Cc + (long)h * Cc;
  for (int c = 0; c < Cc; ++c) atomicAdd(&po[c], w * pl[c]);
}

// per-column sum / sumsq for BatchNorm stats; stats[0..N-1]=sum, stats[N..2N-1]=sumsq
__global__ __launch_bounds__(256)
void colstats_kernel(const float* __restrict__ x, int M, int N, float* __restrict__ stats) {
  int c = threadIdx.x;            // one column per thread, N == 256
  int r0 = blockIdx.x * 128;
  float s = 0.0f, s2 = 0.0f;
  for (int r = r0; r < r0 + 128 && r < M; ++r) {
    float v = x[(long)r * N + c];
    s += v; s2 += v * v;
  }
  atomicAdd(&stats[c], s);
  atomicAdd(&stats[N + c], s2);
}

__global__ void bn_elu_kernel(const float* __restrict__ x, const float* __restrict__ stats,
                              const float* __restrict__ g, const float* __restrict__ b,
                              float* __restrict__ out, int M, int N) {
  long i = (long)blockIdx.x * blockDim.x + threadIdx.x;
  if (i >= (long)M * N) return;
  int c = (int)(i % N);
  float mu  = stats[c] / (float)M;
  float var = stats[N + c] / (float)M - mu * mu;
  float y = (x[i] - mu) * rsqrtf(var + 1e-5f) * g[c] + b[c];
  out[i] = (y > 0.0f) ? y : (expf(y) - 1.0f);
}

__global__ void pool_accum_kernel(const float* __restrict__ h, const int* __restrict__ batch,
                                  float* __restrict__ pooled, float* __restrict__ cnt,
                                  int Nn, int Cc) {
  long i = (long)blockIdx.x * blockDim.x + threadIdx.x;
  if (i >= (long)Nn * Cc) return;
  int node = (int)(i / Cc), c = (int)(i % Cc);
  int b = batch[node];
  atomicAdd(&pooled[(long)b * Cc + c], h[i]);
  if (c == 0) atomicAdd(&cnt[b], 1.0f);
}

__global__ void pool_final_kernel(float* __restrict__ pooled, const float* __restrict__ cnt,
                                  const float* __restrict__ bias, int B, int Cc) {
  long i = (long)blockIdx.x * blockDim.x + threadIdx.x;
  if (i >= (long)B * Cc) return;
  int b = (int)(i / Cc), c = (int)(i % Cc);
  pooled[i] = pooled[i] / fmaxf(cnt[b], 1.0f) + bias[c];
}

__global__ void concat_kernel(const float* __restrict__ d1, const float* __restrict__ d2,
                              const float* __restrict__ t1, const float* __restrict__ t2,
                              const float* __restrict__ p1, const float* __restrict__ p2,
                              const float* __restrict__ ce, const float* __restrict__ tf,
                              float* __restrict__ s, int B) {
  long i = (long)blockIdx.x * blockDim.x + threadIdx.x;
  if (i >= (long)B * 2670) return;
  int c = (int)(i % 2670), b = (int)(i / 2670);
  float v;
  if      (c <  128) v = d1[(long)b * 128  + c];
  else if (c <  256) v = d2[(long)b * 128  + (c - 128)];
  else if (c <  816) v = t1[(long)b * 560  + (c - 256)];
  else if (c < 1376) v = t2[(long)b * 560  + (c - 816)];
  else if (c < 1383) v = p1[(long)b * 7    + (c - 1376)];
  else if (c < 1390) v = p2[(long)b * 7    + (c - 1383)];
  else if (c < 2414) v = ce[(long)b * 1024 + (c - 1390)];
  else               v = tf[(long)b * 256  + (c - 2414)];
  s[i] = v;
}

// LayerNorm(last dim) + exact GELU; one block (256 thr) per row
__global__ __launch_bounds__(256)
void lnorm_gelu_kernel(const float* __restrict__ x, const float* __restrict__ g,
                       const float* __restrict__ b, float* __restrict__ out, int D) {
  __shared__ float sh[512];
  int row = blockIdx.x;
  const float* xr = x + (long)row * D;
  float s = 0.0f, s2 = 0.0f;
  for (int i = threadIdx.x; i < D; i += 256) { float v = xr[i]; s += v; s2 += v * v; }
  sh[threadIdx.x] = s; sh[256 + threadIdx.x] = s2;
  __syncthreads();
  for (int st = 128; st > 0; st >>= 1) {
    if (threadIdx.x < st) {
      sh[threadIdx.x]       += sh[threadIdx.x + st];
      sh[256 + threadIdx.x] += sh[256 + threadIdx.x + st];
    }
    __syncthreads();
  }
  float mu  = sh[0] / (float)D;
  float var = sh[256] / (float)D - mu * mu;
  float rs  = rsqrtf(var + 1e-5f);
  for (int i = threadIdx.x; i < D; i += 256) {
    float y = (xr[i] - mu) * rs * g[i] + b[i];
    out[(long)row * D + i] = 0.5f * y * (1.0f + erff(y * 0.70710678118654752f));
  }
}

// ---------------------------------------------------------------------------
// Orchestration
// ---------------------------------------------------------------------------
extern "C" void kernel_launch(void* const* d_in, const int* in_sizes, int n_in,
                              void* d_out, int out_size, void* d_ws, size_t ws_size,
                              hipStream_t stream) {
  (void)n_in; (void)out_size; (void)ws_size;

  const int Nn = in_sizes[0] / 78;       // 100000
  const int Ed = in_sizes[1] / 2;        // 400000
  const int Bn = in_sizes[6] / 560;      // 2048

  auto F = [&](int i) { return (const float*)d_in[i]; };
  auto I = [&](int i) { return (const int*)d_in[i]; };

  // -------- workspace layout (floats) --------
  float* ws = (float*)d_ws;
  long off = 0;
  auto alloc = [&](long n) { float* p = ws + off; off += n; return p; };
  const long NHC = (long)Nn * 256;
  float* bufH   = alloc(NHC);                 // layer input / BN output
  float* bufXl  = alloc(NHC);                 // xl
  float* bufXr  = alloc(NHC);                 // xr, reused as aggregation acc
  float* logit  = alloc((long)(Ed + Nn) * 4); // edge logits / alpha
  float* segmax = alloc((long)Nn * 4);
  float* segsum = alloc((long)Nn * 4);
  float* cstats = alloc(512);
  float* d1buf  = alloc((long)Bn * 128);
  float* d2buf  = alloc((long)Bn * 128);
  float* cntbuf = alloc(Bn);
  float* sbuf   = alloc((long)Bn * 2670);
  float* tA     = alloc((long)Bn * 1024);
  float* tB     = alloc((long)Bn * 1024);
  float* tC     = alloc((long)Bn * 256);
  float* tD     = alloc((long)Bn * 256);

  auto grid1 = [](long n) { return dim3((unsigned)((n + 255) / 256)); };
  auto gemm = [&](const float* A, const float* W, const float* bias, float* C,
                  int M, int K, int N) {
    dim3 g((N + BN - 1) / BN, (M + BM - 1) / BM);
    gemm_bias_wmma<<<g, 256, 0, stream>>>(A, W, bias, C, M, K, N);
  };

  // -------- one GATv2 layer --------
  auto gat_layer = [&](const float* xin, const int* ei, int fin, int H, int Cc,
                       const float* Wl, const float* bl, const float* Wr, const float* br,
                       const float* att) {
    const int HC = H * Cc;
    const int* src = ei;
    const int* dst = ei + Ed;
    gemm(xin, Wl, bl, bufXl, Nn, fin, HC);
    gemm(xin, Wr, br, bufXr, Nn, fin, HC);
    long tot = (long)(Ed + Nn) * H;
    edge_logit_kernel<<<grid1(tot), 256, 0, stream>>>(bufXl, bufXr, att, src, dst, Ed, Nn, H, Cc, logit);
    fill_kernel<<<grid1((long)Nn * H), 256, 0, stream>>>(segmax, (long)Nn * H, -FLT_MAX);
    fill_kernel<<<grid1((long)Nn * H), 256, 0, stream>>>(segsum, (long)Nn * H, 0.0f);
    edge_max_kernel<<<grid1(tot), 256, 0, stream>>>(logit, dst, Ed, Nn, H, segmax);
    edge_exp_kernel<<<grid1(tot), 256, 0, stream>>>(logit, dst, Ed, Nn, H, segmax, segsum);
    fill_kernel<<<grid1((long)Nn * HC), 256, 0, stream>>>(bufXr, (long)Nn * HC, 0.0f); // xr dead -> acc
    edge_aggregate_kernel<<<grid1(tot), 256, 0, stream>>>(logit, bufXl, src, dst, Ed, Nn, H, Cc, segsum, bufXr);
  };

  auto bn_elu = [&](const float* bng, const float* bnb) {
    fill_kernel<<<grid1(512), 256, 0, stream>>>(cstats, 512, 0.0f);
    colstats_kernel<<<dim3((Nn + 127) / 128), 256, 0, stream>>>(bufXr, Nn, 256, cstats);
    bn_elu_kernel<<<grid1(NHC), 256, 0, stream>>>(bufXr, cstats, bng, bnb, bufH, Nn, 256);
  };

  // -------- encoder: param leaf base G (pytree-sorted: bn1.b,bn1.g,bn2.b,bn2.g,
  //          then per conv: Wl,Wr,att,b,bl,br) --------
  auto encoder = [&](const float* x, const int* ei, const int* batch, int G, float* dpool) {
    gat_layer(x, ei, 78, 4, 64, F(G + 4), F(G + 8), F(G + 5), F(G + 9), F(G + 6));
    bn_elu(F(G + 1), F(G + 0));
    gat_layer(bufH, ei, 256, 4, 64, F(G + 10), F(G + 14), F(G + 11), F(G + 15), F(G + 12));
    bn_elu(F(G + 3), F(G + 2));
    gat_layer(bufH, ei, 256, 1, 128, F(G + 16), F(G + 20), F(G + 17), F(G + 21), F(G + 18));
    fill_kernel<<<grid1((long)Bn * 128), 256, 0, stream>>>(dpool, (long)Bn * 128, 0.0f);
    fill_kernel<<<grid1(Bn), 256, 0, stream>>>(cntbuf, Bn, 0.0f);
    pool_accum_kernel<<<grid1((long)Nn * 128), 256, 0, stream>>>(bufXr, batch, dpool, cntbuf, Nn, 128);
    pool_final_kernel<<<grid1((long)Bn * 128), 256, 0, stream>>>(dpool, cntbuf, F(G + 19), Bn, 128);
  };

  // param leaf indices (12 data inputs, then sorted pytree):
  // attn: 12..19 (k.W,k.b,o.W,o.b,q.W,q.b,v.W,v.b)
  // cls : 20..25 (l1.W,l1.b,l2.W,l2.b,n1.b,n1.g)
  // fus : 26..35 (l1.W,l1.b,l2.W,l2.b,l3.W,l3.b,n1.b,n1.g,n2.b,n2.g)
  // g1  : 36..57, g2: 58..79
  encoder(F(0), I(1), I(2), 36, d1buf);
  encoder(F(3), I(4), I(5), 58, d2buf);

  // fusion input concat [B, 2670]
  concat_kernel<<<grid1((long)Bn * 2670), 256, 0, stream>>>(
      d1buf, d2buf, F(6), F(7), F(8), F(9), F(10), F(11), sbuf, Bn);

  // fusion MLP
  gemm(sbuf, F(26), F(27), tA, Bn, 2670, 1024);
  lnorm_gelu_kernel<<<Bn, 256, 0, stream>>>(tA, F(33), F(32), tB, 1024);
  gemm(tB, F(28), F(29), tA, Bn, 1024, 512);
  lnorm_gelu_kernel<<<Bn, 256, 0, stream>>>(tA, F(35), F(34), tB, 512);
  gemm(tB, F(30), F(31), tC, Bn, 512, 256);            // h [B,256]

  // cross-attn with seq len 1: softmax over single key == 1  =>  r = (h@Wv+bv)@Wo+bo
  gemm(tC, F(18), F(19), tD, Bn, 256, 256);            // v
  gemm(tD, F(14), F(15), tC, Bn, 256, 256);            // r = o-proj

  // classifier
  gemm(tC, F(20), F(21), tA, Bn, 256, 128);
  lnorm_gelu_kernel<<<Bn, 256, 0, stream>>>(tA, F(25), F(24), tB, 128);
  gemm(tB, F(22), F(23), (float*)d_out, Bn, 128, 2);   // [B,2] -> d_out
}
